// MPGA_26611617366258
// MI455X (gfx1250) — matrix operand order repaired and checked by reading, not compiled, not run on hardware
//
#include <hip/hip_runtime.h>
#include <hip/hip_bf16.h>

#define BATCH 8
#define CCH   256
#define NSP   4096   // 64*64 spatial

typedef __attribute__((ext_vector_type(16))) __bf16 v16bf;
typedef __attribute__((ext_vector_type(8)))  float  v8f;

union FragBF {
    v16bf v;
    uint4 q[2];
};

// native bf16 convert (RNE) -- lets the compiler use gfx1250 cvt ops
__device__ __forceinline__ unsigned short f2bf(float f) {
    __bf16 h = (__bf16)f;
    return __builtin_bit_cast(unsigned short, h);
}

// ---- CDNA5 async global->LDS copy (ASYNCcnt-tracked, no VGPR staging) -------
// lds_byte_addr: wave-relative LDS address = low 32 bits of a __shared__
// generic pointer (ISA 10.2: LDS aperture truncates flat addr to addr[31:0]).
// IOFFSET is added to BOTH the LDS and global addresses (ISA 10.7/15.18).
__device__ __forceinline__ void async_b128(unsigned lds_byte_addr,
                                           const void* gsrc, int ioff) {
    asm volatile("global_load_async_to_lds_b128 %0, %1, off offset:%2"
                 :: "v"(lds_byte_addr), "v"(gsrc), "i"(ioff) : "memory");
}
__device__ __forceinline__ void wait_async0() {
#if __has_builtin(__builtin_amdgcn_s_wait_asynccnt)
    __builtin_amdgcn_s_wait_asynccnt(0);
#else
    asm volatile("s_wait_asynccnt 0x0" ::: "memory");
#endif
}

// -----------------------------------------------------------------------------
// Kernel 1: fused 1x1-conv projections (q,k,v) as bf16 WMMA GEMMs.
//   q stored transposed [B][N][C]  (WMMA A-operand friendly)
//   k stored transposed [B][N][C]  (WMMA B-operand friendly for scores)
//   v stored            [B][C][N]  (WMMA B-operand friendly for P*V)
// grid = (N/128, 6, B); y = proj*2 + mhalf
// -----------------------------------------------------------------------------
__global__ __launch_bounds__(256) void proj_kernel(
    const float* __restrict__ queries, const float* __restrict__ keys,
    const float* __restrict__ Wq, const float* __restrict__ bq,
    const float* __restrict__ Wk, const float* __restrict__ bk,
    const float* __restrict__ Wv, const float* __restrict__ bv,
    unsigned short* __restrict__ qT, unsigned short* __restrict__ kT,
    unsigned short* __restrict__ vO)
{
    __shared__ __align__(16) unsigned short sW[128 * 32]; // [m][k] bf16
    __shared__ __align__(16) unsigned short sX[128 * 32]; // [n][k] bf16 (transposed)

    const int b     = blockIdx.z;
    const int proj  = blockIdx.y >> 1;
    const int mhalf = blockIdx.y & 1;
    const int n0    = blockIdx.x * 128;
    const int m0    = mhalf * 128;

    const float* X    = (proj == 0) ? queries : keys;
    const float* W    = (proj == 0) ? Wq : (proj == 1 ? Wk : Wv);
    const float* bias = (proj == 0) ? bq : (proj == 1 ? bk : bv);

    const int tid  = threadIdx.x;
    const int wave = tid >> 5;
    const int lane = tid & 31;
    const int hf   = lane >> 4;   // lane half (k-group select)
    const int lq   = lane & 15;

    v8f acc[8] = {};              // wave: 16 m-rows x 8 n-tiles

    for (int cc = 0; cc < 8; ++cc) {   // K loop over input channels, 32/step
        // stage W[m0..m0+127][cc*32..+31] -> sW[r][k] (k contiguous)
        {
            const int r = tid >> 1, hh = tid & 1;
            const float* wp = W + (size_t)(m0 + r) * CCH + cc * 32 + hh * 16;
            const float4* w4 = (const float4*)wp;
            float4 f0 = w4[0], f1 = w4[1], f2 = w4[2], f3 = w4[3];
            unsigned short* d = &sW[r * 32 + hh * 16];
            d[0]=f2bf(f0.x); d[1]=f2bf(f0.y); d[2]=f2bf(f0.z); d[3]=f2bf(f0.w);
            d[4]=f2bf(f1.x); d[5]=f2bf(f1.y); d[6]=f2bf(f1.z); d[7]=f2bf(f1.w);
            d[8]=f2bf(f2.x); d[9]=f2bf(f2.y); d[10]=f2bf(f2.z); d[11]=f2bf(f2.w);
            d[12]=f2bf(f3.x); d[13]=f2bf(f3.y); d[14]=f2bf(f3.z); d[15]=f2bf(f3.w);
        }
        // stage X chunk transposed: sX[n][c] (c contiguous per n)
        {
            const int c = tid >> 3, nn = (tid & 7) * 16;
            const float* xp = X + ((size_t)b * CCH + cc * 32 + c) * NSP + n0 + nn;
            const float4* x4 = (const float4*)xp;
            float4 f0 = x4[0], f1 = x4[1], f2 = x4[2], f3 = x4[3];
            float fs[16] = {f0.x,f0.y,f0.z,f0.w, f1.x,f1.y,f1.z,f1.w,
                            f2.x,f2.y,f2.z,f2.w, f3.x,f3.y,f3.z,f3.w};
#pragma unroll
            for (int i = 0; i < 16; ++i) sX[(nn + i) * 32 + c] = f2bf(fs[i]);
        }
        __syncthreads();

        FragBF A;   // A: m = wave*16 + lq, k = hf*8 + {0..7} and +16
        {
            const int ao = (wave * 16 + lq) * 32 + hf * 8;
            A.q[0] = *(const uint4*)&sW[ao];
            A.q[1] = *(const uint4*)&sW[ao + 16];
        }
        // preload ALL B fragments first so the ds_loads batch into one clause
        // and the compiler can stagger s_wait_dscnt across the WMMA chain
        FragBF Bf[8];
#pragma unroll
        for (int nt = 0; nt < 8; ++nt) {
            const int bo = (nt * 16 + lq) * 32 + hf * 8;
            Bf[nt].q[0] = *(const uint4*)&sX[bo];
            Bf[nt].q[1] = *(const uint4*)&sX[bo + 16];
        }
#pragma unroll
        for (int nt = 0; nt < 8; ++nt)
            acc[nt] = __builtin_amdgcn_wmma_f32_16x16x32_bf16(
                false, A.v, false, Bf[nt].v, (short)0, acc[nt], false, false);
        __syncthreads();
    }

    // epilogue: + bias, convert bf16, store
    float bb[8];
#pragma unroll
    for (int v = 0; v < 8; ++v) bb[v] = bias[m0 + wave * 16 + hf * 8 + v];

    if (proj < 2) {
        // transposed store [N][C]: per lane the 8 m's are contiguous -> b128
        unsigned short* dst = (proj == 0) ? qT : kT;
#pragma unroll
        for (int nt = 0; nt < 8; ++nt) {
            const int n = n0 + nt * 16 + lq;
            uint4 p;
            p.x = (unsigned)f2bf(acc[nt][0] + bb[0]) | ((unsigned)f2bf(acc[nt][1] + bb[1]) << 16);
            p.y = (unsigned)f2bf(acc[nt][2] + bb[2]) | ((unsigned)f2bf(acc[nt][3] + bb[3]) << 16);
            p.z = (unsigned)f2bf(acc[nt][4] + bb[4]) | ((unsigned)f2bf(acc[nt][5] + bb[5]) << 16);
            p.w = (unsigned)f2bf(acc[nt][6] + bb[6]) | ((unsigned)f2bf(acc[nt][7] + bb[7]) << 16);
            *(uint4*)&dst[((size_t)b * NSP + n) * CCH + m0 + wave * 16 + hf * 8] = p;
        }
    } else {
        // v stored [C][N]; lanes are contiguous in n -> coalesced b16 stores
#pragma unroll
        for (int nt = 0; nt < 8; ++nt)
#pragma unroll
            for (int v = 0; v < 8; ++v) {
                const int m = m0 + wave * 16 + hf * 8 + v;
                vO[((size_t)b * CCH + m) * NSP + n0 + nt * 16 + lq] =
                    f2bf(acc[nt][v] + bb[v]);
            }
    }
}

// -----------------------------------------------------------------------------
// Kernel 2: flash-style masked attention + output blend.
// Double-buffered K/V tiles in LDS filled by async global->LDS DMA; the next
// tile streams in while WMMAs consume the current one.
// Dynamic LDS layout (72 KB total):
//   [0      , 16K)  K buf 0      [16K, 32K)  K buf 1
//   [32K    , 48K)  V buf 0      [48K, 64K)  V buf 1
//   [64K    , 72K)  per-wave P scratch
//   epilogue: transpose tile (16x132 f32) + mask row alias onto K buffers
// -----------------------------------------------------------------------------
#define LDS_K0   0
#define LDS_K1   16384
#define LDS_V0   32768
#define LDS_V1   49152
#define LDS_P    65536
#define LDS_SIZE 73728

__global__ __launch_bounds__(256) void attn_kernel(
    const float* __restrict__ queries, const float* __restrict__ mask,
    const unsigned short* __restrict__ qT, const unsigned short* __restrict__ kT,
    const unsigned short* __restrict__ vO, float* __restrict__ out)
{
    extern __shared__ __align__(16) unsigned char smem[];
    const unsigned smem_lo = (unsigned)(size_t)smem;   // LDS base byte address

    const int b    = blockIdx.z;
    const int q0   = blockIdx.x * 128;
    const int tid  = threadIdx.x;
    const int wave = tid >> 5;
    const int lane = tid & 31;
    const int hf   = lane >> 4;
    const int lq   = lane & 15;

    // staging addresses for this thread (K: row r, 32 bf16 seg; V: row tid)
    const int kr = tid >> 3, kco = (tid & 7) * 32;
    const unsigned kLds = (unsigned)((kr * 256 + kco) * 2);
    const unsigned vLds = (unsigned)(tid * 64);
    const unsigned short* kRow = kT + ((size_t)b * NSP + kr) * CCH + kco;
    const unsigned short* vRow = vO + ((size_t)b * CCH + tid) * NSP;

    // preload Q as A-fragments for all 8 c-chunks (held in registers)
    FragBF QA[8];
    {
        const unsigned short* qp = qT + ((size_t)b * NSP + q0 + wave * 16 + lq) * CCH;
#pragma unroll
        for (int cc = 0; cc < 8; ++cc) {
            QA[cc].q[0] = *(const uint4*)&qp[cc * 32 + hf * 8];
            QA[cc].q[1] = *(const uint4*)&qp[cc * 32 + hf * 8 + 16];
        }
    }

    float mq[8];
#pragma unroll
    for (int v = 0; v < 8; ++v)
        mq[v] = mask[(size_t)b * NSP + q0 + wave * 16 + hf * 8 + v];

    float m_s[8], l_s[8];
#pragma unroll
    for (int v = 0; v < 8; ++v) { m_s[v] = -3.0e38f; l_s[v] = 0.0f; }
    v8f O[16] = {};   // 16 q-rows x 256 channels, f32

    unsigned short* sPw = (unsigned short*)(smem + LDS_P) + wave * 16 * 32;

    // async-prefetch tile 0 into buffer 0
    {
        const unsigned kb = smem_lo + LDS_K0 + kLds;
        const unsigned vb = smem_lo + LDS_V0 + vLds;
        async_b128(kb, kRow, 0);  async_b128(kb, kRow, 16);
        async_b128(kb, kRow, 32); async_b128(kb, kRow, 48);
        async_b128(vb, vRow, 0);  async_b128(vb, vRow, 16);
        async_b128(vb, vRow, 32); async_b128(vb, vRow, 48);
    }

    for (int kt = 0; kt < NSP / 32; ++kt) {
        const int cur = kt & 1;
        wait_async0();      // this wave's tile-kt DMA done
        __syncthreads();    // everyone's tile-kt visible; prev buf free

        if (kt + 1 < NSP / 32) {   // prefetch next tile into the other buffer
            const int k1 = (kt + 1) * 32;
            const unsigned kb = smem_lo + (cur ? LDS_K0 : LDS_K1) + kLds;
            const unsigned vb = smem_lo + (cur ? LDS_V0 : LDS_V1) + vLds;
            const unsigned short* kg = kRow + (size_t)k1 * CCH;
            const unsigned short* vg = vRow + k1;
            async_b128(kb, kg, 0);  async_b128(kb, kg, 16);
            async_b128(kb, kg, 32); async_b128(kb, kg, 48);
            async_b128(vb, vg, 0);  async_b128(vb, vg, 16);
            async_b128(vb, vg, 32); async_b128(vb, vg, 48);
        }
        // select current buffers by integer offset (no LDS pointer arrays!)
        const unsigned short* sK =
            (const unsigned short*)(smem + (cur ? LDS_K1 : LDS_K0));
        const unsigned short* sV =
            (const unsigned short*)(smem + (cur ? LDS_V1 : LDS_V0));

        // scores S = Q x K^T for 16 q-rows x 32 keys (2 tiles, K=256)
        v8f S[2] = {};
#pragma unroll
        for (int j = 0; j < 2; ++j) {
            FragBF Bf[8];   // batch the ds_loads ahead of the WMMA chain
#pragma unroll
            for (int cc = 0; cc < 8; ++cc) {
                const int bo = (j * 16 + lq) * 256 + cc * 32 + hf * 8;
                Bf[cc].q[0] = *(const uint4*)&sK[bo];
                Bf[cc].q[1] = *(const uint4*)&sK[bo + 16];
            }
#pragma unroll
            for (int cc = 0; cc < 8; ++cc)
                S[j] = __builtin_amdgcn_wmma_f32_16x16x32_bf16(
                    false, QA[cc].v, false, Bf[cc].v, (short)0, S[j], false, false);
        }

        // online softmax; logits = mask[q] * score (per reference)
        float p0[8], p1[8], alpha[8];
#pragma unroll
        for (int v = 0; v < 8; ++v) {
            float s0 = mq[v] * S[0][v];
            float s1 = mq[v] * S[1][v];
            float mx = fmaxf(s0, s1);
            mx = fmaxf(mx, __shfl_xor(mx, 1, 32));
            mx = fmaxf(mx, __shfl_xor(mx, 2, 32));
            mx = fmaxf(mx, __shfl_xor(mx, 4, 32));
            mx = fmaxf(mx, __shfl_xor(mx, 8, 32));
            const float nm = fmaxf(m_s[v], mx);
            alpha[v] = __expf(m_s[v] - nm);
            m_s[v] = nm;
            p0[v] = __expf(s0 - nm);
            p1[v] = __expf(s1 - nm);
            float rs = p0[v] + p1[v];
            rs += __shfl_xor(rs, 1, 32);
            rs += __shfl_xor(rs, 2, 32);
            rs += __shfl_xor(rs, 4, 32);
            rs += __shfl_xor(rs, 8, 32);
            l_s[v] = l_s[v] * alpha[v] + rs;
        }
#pragma unroll
        for (int ct = 0; ct < 16; ++ct)
#pragma unroll
            for (int v = 0; v < 8; ++v) O[ct][v] *= alpha[v];

        // C/D layout -> A layout for P via per-wave LDS round trip
#pragma unroll
        for (int v = 0; v < 8; ++v) {
            sPw[(hf * 8 + v) * 32 + lq]      = f2bf(p0[v]);
            sPw[(hf * 8 + v) * 32 + 16 + lq] = f2bf(p1[v]);
        }
        FragBF PA;
        PA.q[0] = *(const uint4*)&sPw[lq * 32 + hf * 8];
        PA.q[1] = *(const uint4*)&sPw[lq * 32 + hf * 8 + 16];

        // O += P x V  (16 channel tiles)
#pragma unroll
        for (int ct = 0; ct < 16; ++ct) {
            FragBF Bf;  // n = channel ct*16+lq, k-dim = key
            const int bo = (ct * 16 + lq) * 32 + hf * 8;
            Bf.q[0] = *(const uint4*)&sV[bo];
            Bf.q[1] = *(const uint4*)&sV[bo + 16];
            O[ct] = __builtin_amdgcn_wmma_f32_16x16x32_bf16(
                false, PA.v, false, Bf.v, (short)0, O[ct], false, false);
        }
    }
    __syncthreads();   // all compute done; K buffers reusable for transpose

    // ---- coalesced epilogue: transpose each 16-channel tile through LDS ----
    float* sT    = (float*)smem;            // [16][132] f32, padded stride
    float* sMask = (float*)(smem + 9216);   // [128] f32
    if (tid < 128) sMask[tid] = mask[(size_t)b * NSP + q0 + tid];

    float inv[8];
#pragma unroll
    for (int v = 0; v < 8; ++v) inv[v] = 1.0f / l_s[v];

    const int cl   = tid >> 4;          // channel row within tile (0..15)
    const int qseg = (tid & 15) * 8;    // 8-query segment (0..120)

    for (int ct = 0; ct < 16; ++ct) {
#pragma unroll
        for (int v = 0; v < 8; ++v)
            sT[lq * 132 + wave * 16 + hf * 8 + v] = O[ct][v] * inv[v];
        __syncthreads();

        const int c = ct * 16 + cl;
        const size_t base = ((size_t)b * CCH + c) * NSP + q0 + qseg;
        const float4* q4 = (const float4*)(queries + base);
        float4 qa = q4[0], qb4 = q4[1];
        float o[8];
#pragma unroll
        for (int i = 0; i < 8; ++i) {
            const float qv = (i < 4) ? ((const float*)&qa)[i] : ((const float*)&qb4)[i - 4];
            const float mv = sMask[qseg + i];
            o[i] = qv * mv + (1.0f - mv) * sT[cl * 132 + qseg + i];
        }
        float4* o4 = (float4*)(out + base);
        o4[0] = make_float4(o[0], o[1], o[2], o[3]);
        o4[1] = make_float4(o[4], o[5], o[6], o[7]);
        __syncthreads();
    }
}

// -----------------------------------------------------------------------------
extern "C" void kernel_launch(void* const* d_in, const int* in_sizes, int n_in,
                              void* d_out, int out_size, void* d_ws, size_t ws_size,
                              hipStream_t stream) {
    const float* queries = (const float*)d_in[0];
    const float* keys    = (const float*)d_in[1];
    const float* mask    = (const float*)d_in[2];
    const float* Wq      = (const float*)d_in[3];
    const float* bq      = (const float*)d_in[4];
    const float* Wk      = (const float*)d_in[5];
    const float* bk      = (const float*)d_in[6];
    const float* Wv      = (const float*)d_in[7];
    const float* bv      = (const float*)d_in[8];

    const size_t planeElems = (size_t)BATCH * NSP * CCH;   // 8M bf16 = 16 MB each
    unsigned short* qT = (unsigned short*)d_ws;
    unsigned short* kT = qT + planeElems;
    unsigned short* vB = kT + planeElems;

    proj_kernel<<<dim3(NSP / 128, 6, BATCH), 256, 0, stream>>>(
        queries, keys, Wq, bq, Wk, bk, Wv, bv, qT, kT, vB);
    attn_kernel<<<dim3(NSP / 128, 1, BATCH), 256, LDS_SIZE, stream>>>(
        queries, mask, qT, kT, vB, (float*)d_out);
}
